// GraphSAGE_60086592471214
// MI455X (gfx1250) — compile-verified, matrix operand order checked by reading, not run on hardware
//
#include <hip/hip_runtime.h>

// ---------------------------------------------------------------------------
// GraphSAGE (2-layer, mean aggr) for MI455X / gfx1250.
//  - bf16 WMMA (v_wmma_f32_16x16x32_bf16) for all node-feature GEMMs
//  - weights staged to LDS via CDNA5 async copy (global_load_async_to_lds_b128)
//  - f32 global atomics for edge aggregation, prefetched edge streams
//  - fused bias/ReLU (layer 1) and bias/log-softmax (layer 2)
// ---------------------------------------------------------------------------

typedef __attribute__((ext_vector_type(16))) __bf16 v16bf;
typedef __attribute__((ext_vector_type(8)))  float  v8f;

#define NODES   100000
#define EDGES   1000000
#define DIN     128
#define DHID    128
#define DOUT    64
#define NTILES  (NODES / 16)    // 6250 exactly
#define NJOBS1  (NODES / 32)    // 3125 exactly (M=32 blocking in layer 1)

__device__ __forceinline__ __bf16 f2bf(float x) { return (__bf16)x; }

// 32-bit LDS offset of a generic pointer known to point into LDS.
__device__ __forceinline__ unsigned lds_addr_of(const void* p) {
    return (unsigned)(size_t)(__attribute__((address_space(3))) const void*)p;
}

// CDNA5 async copy: 16 bytes per lane, global -> LDS, tracked by ASYNCcnt.
__device__ __forceinline__ void async_cp16(unsigned lds_addr, const void* gp) {
    asm volatile("global_load_async_to_lds_b128 %0, %1, off"
                 :: "v"(lds_addr), "v"(gp) : "memory");
}
__device__ __forceinline__ void wait_async0() {
    asm volatile("s_wait_asynccnt 0" ::: "memory");
}

// ---------------------------------------------------------------------------
// Zero workspace (float4 granularity, grid-stride)
// ---------------------------------------------------------------------------
__global__ __launch_bounds__(256) void zero_kernel(float4* p, size_t n4) {
    size_t i = (size_t)blockIdx.x * blockDim.x + threadIdx.x;
    size_t stride = (size_t)gridDim.x * blockDim.x;
    float4 z; z.x = 0.f; z.y = 0.f; z.z = 0.f; z.w = 0.f;
    for (; i < n4; i += stride) p[i] = z;
}

// ---------------------------------------------------------------------------
// Weights f32 [K x Nout] row-major  ->  bf16 W^T [Nout x K] row-major.
// Each output-column's K-vector is contiguous, so B-fragments are two
// contiguous 16B chunks per lane (same addressing pattern as A-fragments).
// ---------------------------------------------------------------------------
__global__ __launch_bounds__(256) void prep_weights(
    const float* __restrict__ Wl1, const float* __restrict__ Wr1,
    const float* __restrict__ Wl2, const float* __restrict__ Wr2,
    __bf16* __restrict__ wtl1, __bf16* __restrict__ wtr1,
    __bf16* __restrict__ wtl2, __bf16* __restrict__ wtr2) {
    int idx = blockIdx.x * blockDim.x + threadIdx.x;
    if (idx < DIN * DHID) {                 // 128x128
        int k = idx >> 7, n = idx & 127;
        wtl1[n * DIN + k] = f2bf(Wl1[idx]);
        wtr1[n * DIN + k] = f2bf(Wr1[idx]);
    }
    if (idx < DHID * DOUT) {                // 128x64
        int k = idx / DOUT, n = idx % DOUT;
        wtl2[n * DHID + k] = f2bf(Wl2[idx]);
        wtr2[n * DHID + k] = f2bf(Wr2[idx]);
    }
}

// ---------------------------------------------------------------------------
// Edge scatter: one wave per edge (grid-stride). Lane l gathers 4 consecutive
// features (float4, coalesced 512B row read) and does 4 global_atomic_add_f32
// into msg[dst]. Lane 0 accumulates the in-degree. Edge streams prefetched.
// ---------------------------------------------------------------------------
__global__ __launch_bounds__(256) void scatter_kernel(
    const float* __restrict__ feat, const int* __restrict__ src,
    const int* __restrict__ dst, float* msg, float* cnt, int nedges) {
    int gid    = blockIdx.x * blockDim.x + threadIdx.x;
    int wave   = gid >> 5;
    int lane   = gid & 31;
    int nwaves = (gridDim.x * blockDim.x) >> 5;
    for (int e = wave; e < nedges; e += nwaves) {
        __builtin_prefetch(src + e + nwaves, 0, 1);   // global_prefetch_b8
        __builtin_prefetch(dst + e + nwaves, 0, 1);
        int s = src[e];
        int d = dst[e];
        float4 v = ((const float4*)(feat + (size_t)s * 128))[lane];
        float* mp = msg + (size_t)d * 128 + lane * 4;
        atomicAdd(mp + 0, v.x);
        atomicAdd(mp + 1, v.y);
        atomicAdd(mp + 2, v.z);
        atomicAdd(mp + 3, v.w);
        if (lane == 0) atomicAdd(cnt + d, 1.0f);
    }
}

// ---------------------------------------------------------------------------
// A-fragment builder: bf16 16x32 A layout (lanes 0-15: row M=lane, K chunks
// {0..7,16..23}; lanes 16-31: row M=lane-16, K chunks {8..15,24..31}).
// Loads from an f32 row (128 long) with a per-row scale.
// ---------------------------------------------------------------------------
__device__ __forceinline__ v16bf make_frag_f32(const float* row, int kstep,
                                               int hi, float scale) {
    int o0 = kstep * 32 + hi * 8;
    v16bf f;
#pragma unroll
    for (int i = 0; i < 8; ++i) {
        f[i]     = f2bf(row[o0 + i]      * scale);
        f[i + 8] = f2bf(row[o0 + 16 + i] * scale);
    }
    return f;
}

// B-fragment from a bf16 W^T row (K contiguous; here sourced from LDS).
__device__ __forceinline__ v16bf make_frag_bf16(const __bf16* row, int kstep,
                                                int hi) {
    int o0 = kstep * 32 + hi * 8;
    v16bf f;
#pragma unroll
    for (int i = 0; i < 8; ++i) {
        f[i]     = row[o0 + i];
        f[i + 8] = row[o0 + 16 + i];
    }
    return f;
}

#define WMMA_BF16(A, B, C) \
    __builtin_amdgcn_wmma_f32_16x16x32_bf16(false, (A), false, (B), (short)0, \
                                            (C), false, false)

// ---------------------------------------------------------------------------
// Layer 1: one wave per 32-row block (two 16-row tiles sharing B fragments).
//   h = relu( (msg/deg) @ W_l1 + b1 + x @ W_r1 ),  128 -> 128
// Weights (64 KB bf16) async-staged into LDS once per workgroup; inner-loop
// B fragments come from ds_load. 16 WMMAs per column tile, 128 per wave.
// Writes h in place over msg (block-disjoint rows -> safe).
// ---------------------------------------------------------------------------
__global__ __launch_bounds__(256) void gemm1_kernel(
    const float* msg, const float* __restrict__ x,
    const __bf16* __restrict__ wt,      // wtl1 | wtr1, contiguous 64 KB
    const float* __restrict__ bias, const float* __restrict__ cnt,
    float* hout, int njobs) {
    __shared__ __bf16 smem[2 * DHID * DIN];          // 64 KB: W_l^T | W_r^T
    {
        unsigned sbase = lds_addr_of(smem);
        const char* g = (const char*)wt;
        for (int c = threadIdx.x; c < (2 * DHID * DIN * 2) / 16; c += 256)
            async_cp16(sbase + c * 16, g + c * 16);
        wait_async0();
    }
    __syncthreads();

    int wave = (blockIdx.x * blockDim.x + threadIdx.x) >> 5;
    if (wave >= njobs) return;          // wave-uniform: EXEC stays all-ones
    int lane = threadIdx.x & 31;
    int m16  = lane & 15;
    int hi   = lane >> 4;
    int row0 = wave * 32;

    v16bf a_agg[2][4], a_x[2][4];
#pragma unroll
    for (int t = 0; t < 2; ++t) {
        int r = row0 + t * 16 + m16;
        float inv = 1.0f / fmaxf(cnt[r], 1.0f);
        const float* mrow = msg + (size_t)r * DHID;
        const float* xrow = x   + (size_t)r * DIN;
#pragma unroll
        for (int k = 0; k < 4; ++k) {
            a_agg[t][k] = make_frag_f32(mrow, k, hi, inv);
            a_x[t][k]   = make_frag_f32(xrow, k, hi, 1.0f);
        }
    }

    for (int j = 0; j < 8; ++j) {                    // DHID / 16 column tiles
        float bv = bias[j * 16 + m16];
        v8f acc0, acc1;
#pragma unroll
        for (int r = 0; r < 8; ++r) { acc0[r] = bv; acc1[r] = bv; }
        const __bf16* pl = smem + (size_t)(j * 16 + m16) * DIN;
        const __bf16* pr = smem + DHID * DIN + (size_t)(j * 16 + m16) * DIN;
#pragma unroll
        for (int k = 0; k < 4; ++k) {
            v16bf bl = make_frag_bf16(pl, k, hi);    // shared by both tiles
            v16bf br = make_frag_bf16(pr, k, hi);
            acc0 = WMMA_BF16(a_agg[0][k], bl, acc0);
            acc0 = WMMA_BF16(a_x[0][k],   br, acc0);
            acc1 = WMMA_BF16(a_agg[1][k], bl, acc1);
            acc1 = WMMA_BF16(a_x[1][k],   br, acc1);
        }
#pragma unroll
        for (int r = 0; r < 8; ++r) {                // C layout: row r (+8 hi)
            float v0 = fmaxf(acc0[r], 0.0f);         // ReLU
            float v1 = fmaxf(acc1[r], 0.0f);
            hout[(size_t)(row0 +      r + hi * 8) * DHID + j * 16 + m16] = v0;
            hout[(size_t)(row0 + 16 + r + hi * 8) * DHID + j * 16 + m16] = v1;
        }
    }
}

// ---------------------------------------------------------------------------
// Layer 2 + log_softmax: one wave per 16-row tile, 4 col tiles (DOUT=64).
//   logits = (msg2/deg) @ W_l2 + b2 + h @ W_r2
//   out    = logits - (max + log(sum(exp(logits - max))))   per row
// Weights (32 KB bf16) async-staged into LDS. Row-wise reduction done on the
// WMMA C layout via __shfl_xor width 16 within each half-wave.
// ---------------------------------------------------------------------------
__global__ __launch_bounds__(256) void gemm2_kernel(
    const float* __restrict__ msg2, const float* __restrict__ h,
    const __bf16* __restrict__ wt,      // wtl2 | wtr2, contiguous 32 KB
    const float* __restrict__ bias, const float* __restrict__ cnt,
    float* __restrict__ out, int ntiles) {
    __shared__ __bf16 smem[2 * DOUT * DHID];         // 32 KB: W_l^T | W_r^T
    {
        unsigned sbase = lds_addr_of(smem);
        const char* g = (const char*)wt;
        for (int c = threadIdx.x; c < (2 * DOUT * DHID * 2) / 16; c += 256)
            async_cp16(sbase + c * 16, g + c * 16);
        wait_async0();
    }
    __syncthreads();

    int wave = (blockIdx.x * blockDim.x + threadIdx.x) >> 5;
    if (wave >= ntiles) return;
    int lane = threadIdx.x & 31;
    int m16  = lane & 15;
    int hi   = lane >> 4;
    int row0 = wave * 16;

    float inv = 1.0f / fmaxf(cnt[row0 + m16], 1.0f);
    const float* mrow = msg2 + (size_t)(row0 + m16) * DHID;
    const float* hrow = h    + (size_t)(row0 + m16) * DHID;

    v16bf a_agg[4], a_h[4];
#pragma unroll
    for (int k = 0; k < 4; ++k) {
        a_agg[k] = make_frag_f32(mrow, k, hi, inv);
        a_h[k]   = make_frag_f32(hrow, k, hi, 1.0f);
    }

    v8f acc[4];
#pragma unroll
    for (int t = 0; t < 4; ++t) {                    // DOUT / 16 column tiles
        float bv = bias[t * 16 + m16];
        v8f a;
#pragma unroll
        for (int r = 0; r < 8; ++r) a[r] = bv;
        const __bf16* pl = smem + (size_t)(t * 16 + m16) * DHID;
        const __bf16* pr = smem + DOUT * DHID + (size_t)(t * 16 + m16) * DHID;
#pragma unroll
        for (int k = 0; k < 4; ++k) {
            v16bf bl = make_frag_bf16(pl, k, hi);
            v16bf br = make_frag_bf16(pr, k, hi);
            a = WMMA_BF16(a_agg[k], bl, a);
            a = WMMA_BF16(a_h[k],   br, a);
        }
        acc[t] = a;
    }

#pragma unroll
    for (int r = 0; r < 8; ++r) {
        float mx = fmaxf(fmaxf(acc[0][r], acc[1][r]),
                         fmaxf(acc[2][r], acc[3][r]));
#pragma unroll
        for (int off = 1; off < 16; off <<= 1)
            mx = fmaxf(mx, __shfl_xor(mx, off, 16));
        float s = 0.f;
#pragma unroll
        for (int t = 0; t < 4; ++t) s += __expf(acc[t][r] - mx);
#pragma unroll
        for (int off = 1; off < 16; off <<= 1)
            s += __shfl_xor(s, off, 16);
        float lse = mx + __logf(s);
        size_t base = (size_t)(row0 + r + hi * 8) * DOUT;
#pragma unroll
        for (int t = 0; t < 4; ++t)
            out[base + t * 16 + m16] = acc[t][r] - lse;
    }
}

// ---------------------------------------------------------------------------
// Launch
// ---------------------------------------------------------------------------
extern "C" void kernel_launch(void* const* d_in, const int* in_sizes, int n_in,
                              void* d_out, int out_size, void* d_ws, size_t ws_size,
                              hipStream_t stream) {
    const float* x   = (const float*)d_in[0];
    const int*   ei1 = (const int*)d_in[1];   // [2, E]: src row 0, dst row 1
    const int*   ei2 = (const int*)d_in[2];
    const float* Wl1 = (const float*)d_in[3];
    const float* bl1 = (const float*)d_in[4];
    const float* Wr1 = (const float*)d_in[5];
    const float* Wl2 = (const float*)d_in[6];
    const float* bl2 = (const float*)d_in[7];
    const float* Wr2 = (const float*)d_in[8];
    float* out = (float*)d_out;

    // Workspace layout (~103.3 MB):
    //   bufA: msg1 then h (in place)   N*128 f32
    //   bufB: msg2                     N*128 f32
    //   cnt1, cnt2:                    N f32 each
    //   bf16 W^T blocks (fragment-friendly order, pairs contiguous for LDS
    //   staging: wtl1|wtr1 = 64 KB, wtl2|wtr2 = 32 KB)
    float* bufA = (float*)d_ws;
    float* bufB = bufA + (size_t)NODES * DHID;
    float* cnt1 = bufB + (size_t)NODES * DHID;
    float* cnt2 = cnt1 + NODES;
    __bf16* wtl1 = (__bf16*)(cnt2 + NODES);
    __bf16* wtr1 = wtl1 + DIN * DHID;
    __bf16* wtl2 = wtr1 + DIN * DHID;
    __bf16* wtr2 = wtl2 + DHID * DOUT;

    size_t n4 = ((size_t)NODES * DHID * 2 + 2 * (size_t)NODES) / 4;
    zero_kernel<<<4096, 256, 0, stream>>>((float4*)d_ws, n4);
    prep_weights<<<64, 256, 0, stream>>>(Wl1, Wr1, Wl2, Wr2,
                                         wtl1, wtr1, wtl2, wtr2);
    // Layer 1 aggregation: msg1 += x[src1] @ dst1, cnt1 = in-degree
    scatter_kernel<<<4096, 256, 0, stream>>>(x, ei1, ei1 + EDGES,
                                             bufA, cnt1, EDGES);
    gemm1_kernel<<<(NJOBS1 * 32 + 255) / 256, 256, 0, stream>>>(
        bufA, x, wtl1, bl1, cnt1, bufA, NJOBS1);
    // Layer 2 aggregation over h
    scatter_kernel<<<4096, 256, 0, stream>>>(bufA, ei2, ei2 + EDGES,
                                             bufB, cnt2, EDGES);
    gemm2_kernel<<<(NTILES + 7) / 8, 256, 0, stream>>>(
        bufB, bufA, wtl2, bl2, cnt2, out, NTILES);
}
